// Conv1D_LowRankLSTM_89876485636763
// MI455X (gfx1250) — matrix operand order repaired and checked by reading, not compiled, not run on hardware
//
#include <hip/hip_runtime.h>

// ---------------------------------------------------------------------------
// Problem constants (from the reference)
// ---------------------------------------------------------------------------
#define B_DIM 256
#define S_DIM 1024
#define CIN0  64
#define CMID  32
#define H_DIM 128
#define G_DIM 512      // 4*H
#define BT    16       // batch tile per workgroup in the LSTM kernel
#define NCLS  12
#define BN_EPS 1e-5f

typedef __attribute__((ext_vector_type(16))) __bf16        v16bf;
typedef __attribute__((ext_vector_type(8)))  float         v8f;
typedef __attribute__((ext_vector_type(8)))  unsigned int  v8u;

__device__ __forceinline__ unsigned short f2bf(float f) {
    unsigned int u = __float_as_uint(f);
    unsigned int r = u + 0x7FFFu + ((u >> 16) & 1u);   // round-to-nearest-even
    return (unsigned short)(r >> 16);
}
__device__ __forceinline__ float sigmoidf(float x) {
    return 1.0f / (1.0f + __expf(-x));
}

// ---------------------------------------------------------------------------
// Fused depthwise(k=5,pad=2) + pointwise(1x1) conv, [B,S,cin] -> [B,S,32].
// Also produces per-block channel sum / sum-of-squares partials for BN.
// ---------------------------------------------------------------------------
__global__ void dwpw_kernel(const float* __restrict__ in,
                            const float* __restrict__ dwk,   // [cin,1,5]
                            const float* __restrict__ dwb,   // [cin]
                            const float* __restrict__ pwk,   // [32,cin,1]
                            const float* __restrict__ pwb,   // [32]
                            float* __restrict__ y,           // [B,S,32]
                            float* __restrict__ partial,     // [nblk][64]
                            int cin, int S)
{
    __shared__ float sPW[32 * 64];
    __shared__ float sDW[64 * 5];
    __shared__ float sDB[64];
    __shared__ float sPB[32];
    __shared__ float sSum[32];
    __shared__ float sSq[32];

    const int t = threadIdx.x;
    for (int i = t; i < 32 * cin; i += 256) sPW[i] = pwk[i];
    for (int i = t; i < cin * 5;  i += 256) sDW[i] = dwk[i];
    if (t < cin) sDB[t] = dwb[t];
    if (t < 32) { sPB[t] = pwb[t]; sSum[t] = 0.0f; sSq[t] = 0.0f; }
    __syncthreads();

    const int gid = blockIdx.x * 256 + t;   // flat (b,s); grid sized exactly
    const int b = gid / S;
    const int s = gid % S;

    float acc[32];
    #pragma unroll
    for (int co = 0; co < 32; ++co) acc[co] = sPB[co];

    for (int c = 0; c < cin; ++c) {
        float d = sDB[c];
        #pragma unroll
        for (int k = 0; k < 5; ++k) {
            int s2 = s + k - 2;
            if (s2 >= 0 && s2 < S)
                d += in[(b * S + s2) * cin + c] * sDW[c * 5 + k];
        }
        #pragma unroll
        for (int co = 0; co < 32; ++co) acc[co] += sPW[co * cin + c] * d;
    }

    float* yrow = y + (size_t)gid * 32;
    #pragma unroll
    for (int co = 0; co < 32; ++co) {
        yrow[co] = acc[co];
        atomicAdd(&sSum[co], acc[co]);          // LDS ds_add_f32
        atomicAdd(&sSq[co],  acc[co] * acc[co]);
    }
    __syncthreads();
    if (t < 32) {
        partial[blockIdx.x * 64 + t]      = sSum[t];
        partial[blockIdx.x * 64 + 32 + t] = sSq[t];
    }
}

// Fixed-order reduction of block partials -> fused scale/shift (BN folded).
__global__ void bnstats_kernel(const float* __restrict__ partial, int nblk,
                               const float* __restrict__ gamma,
                               const float* __restrict__ beta,
                               float* __restrict__ scsh, float inv_n)
{
    int c = threadIdx.x;
    if (c < 32) {
        float s = 0.0f, q = 0.0f;
        for (int blk = 0; blk < nblk; ++blk) {
            s += partial[blk * 64 + c];
            q += partial[blk * 64 + 32 + c];
        }
        float mean = s * inv_n;
        float var  = q * inv_n - mean * mean;
        float sc   = gamma[c] * rsqrtf(var + BN_EPS);
        scsh[c]      = sc;
        scsh[32 + c] = beta[c] - mean * sc;
    }
}

// Apply BN + ReLU; optionally emit bf16 features for the WMMA LSTM stage.
__global__ void bnapply_kernel(const float* __restrict__ y,
                               const float* __restrict__ scsh,
                               float* __restrict__ act,
                               unsigned short* __restrict__ actbf,
                               int n)
{
    int i = blockIdx.x * 256 + threadIdx.x;
    if (i < n) {
        int c = i & 31;
        float v = y[i] * scsh[c] + scsh[32 + c];
        v = v > 0.0f ? v : 0.0f;
        act[i] = v;
        if (actbf) actbf[i] = f2bf(v);
    }
}

// ---------------------------------------------------------------------------
// Reconstruct low-rank LSTM weights W = U@V, store TRANSPOSED [N=512][K] bf16
// (K-contiguous so B-fragments load one dword per 2 K-values), and fuse biases.
// ---------------------------------------------------------------------------
__global__ void weight_prep_kernel(const float* __restrict__ U_i1, const float* __restrict__ V_i1,
                                   const float* __restrict__ U_h1, const float* __restrict__ V_h1,
                                   const float* __restrict__ bi1,  const float* __restrict__ bh1,
                                   const float* __restrict__ U_i2, const float* __restrict__ V_i2,
                                   const float* __restrict__ U_h2, const float* __restrict__ V_h2,
                                   const float* __restrict__ bi2,  const float* __restrict__ bh2,
                                   unsigned short* __restrict__ Wti1, unsigned short* __restrict__ Wth1,
                                   unsigned short* __restrict__ Wti2, unsigned short* __restrict__ Wth2,
                                   float* __restrict__ b1, float* __restrict__ b2)
{
    int idx = blockIdx.x * 256 + threadIdx.x;
    const int NI1 = G_DIM * CMID;      // 512*32
    const int NH  = G_DIM * H_DIM;     // 512*128
    if (idx < NI1) {
        int n = idx / CMID, c = idx % CMID;
        float s = 0.0f;
        for (int r = 0; r < 16; ++r) s += U_i1[c * 16 + r] * V_i1[r * G_DIM + n];
        Wti1[n * CMID + c] = f2bf(s);
        return;
    }
    idx -= NI1;
    if (idx < NH) {
        int n = idx / H_DIM, h = idx % H_DIM;
        float s = 0.0f;
        for (int r = 0; r < 16; ++r) s += U_h1[h * 16 + r] * V_h1[r * G_DIM + n];
        Wth1[n * H_DIM + h] = f2bf(s);
        return;
    }
    idx -= NH;
    if (idx < NH) {
        int n = idx / H_DIM, h = idx % H_DIM;
        float s = 0.0f;
        for (int r = 0; r < 16; ++r) s += U_i2[h * 16 + r] * V_i2[r * G_DIM + n];
        Wti2[n * H_DIM + h] = f2bf(s);
        return;
    }
    idx -= NH;
    if (idx < NH) {
        int n = idx / H_DIM, h = idx % H_DIM;
        float s = 0.0f;
        for (int r = 0; r < 16; ++r) s += U_h2[h * 16 + r] * V_h2[r * G_DIM + n];
        Wth2[n * H_DIM + h] = f2bf(s);
        return;
    }
    idx -= NH;
    if (idx < G_DIM) {
        b1[idx] = bi1[idx] + bh1[idx];
        b2[idx] = bi2[idx] + bh2[idx];
    }
}

// ---------------------------------------------------------------------------
// Persistent 2-layer LSTM + final FC. One workgroup (8 wave32) per 16-row
// batch tile; 1024 sequential steps; W_h1/W_h2 resident in 256 KB of LDS;
// W_i1/W_i2 streamed from L2; biases in LDS; all GEMM tiles via
// v_wmma_f32_16x16x32_bf16; cell state c lives in VGPRs for all 1024 steps.
// ---------------------------------------------------------------------------
__global__ void __launch_bounds__(256)
lstm_kernel(const unsigned short* __restrict__ feats, // [B,S,32] bf16
            const unsigned short* __restrict__ Wti1,  // [512][32]  (streamed)
            const unsigned short* __restrict__ Wth1,  // [512][128] (-> LDS)
            const unsigned short* __restrict__ Wti2,  // [512][128] (streamed)
            const unsigned short* __restrict__ Wth2,  // [512][128] (-> LDS)
            const float* __restrict__ b1,
            const float* __restrict__ b2,
            const float* __restrict__ fcw,            // [128,12]
            const float* __restrict__ fcb,            // [12]
            float* __restrict__ out)                  // [256,12]
{
    extern __shared__ unsigned char smem_raw[];
    unsigned short* sWh1 = (unsigned short*)smem_raw;                 // 512*128 bf16 (128 KB)
    unsigned short* sWh2 = sWh1 + G_DIM * H_DIM;                      // 128 KB
    float*          sG   = (float*)(sWh2 + G_DIM * H_DIM);            // 16*512 f32 (32 KB)
    unsigned short* sH1  = (unsigned short*)(sG + BT * G_DIM);        // 16*128 bf16
    unsigned short* sH2  = sH1 + BT * H_DIM;
    float*          sB1  = (float*)(sH2 + BT * H_DIM);                // 512 f32
    float*          sB2  = sB1 + G_DIM;                               // 512 f32

    const int tid  = threadIdx.x;
    const int lane = tid & 31;
    const int wv   = tid >> 5;          // wave 0..7; each owns 4 N-tiles
    const int mlan = lane & 15;
    const int half = lane >> 4;
    const int m0   = blockIdx.x * BT;   // batch tile base row

    // Stage recurrent weights + biases into LDS (once for all 1024 steps).
    {
        const uint4* a  = (const uint4*)Wth1;
        const uint4* b_ = (const uint4*)Wth2;
        uint4* da = (uint4*)sWh1;
        uint4* db = (uint4*)sWh2;
        for (int i = tid; i < G_DIM * H_DIM / 8; i += blockDim.x) { da[i] = a[i]; db[i] = b_[i]; }
    }
    for (int i = tid; i < G_DIM; i += blockDim.x) { sB1[i] = b1[i]; sB2[i] = b2[i]; }
    for (int i = tid; i < BT * H_DIM; i += blockDim.x) { sH1[i] = 0; sH2[i] = 0; }
    float c1r[8], c2r[8], h2f[8];
    #pragma unroll
    for (int j = 0; j < 8; ++j) { c1r[j] = 0.0f; c2r[j] = 0.0f; h2f[j] = 0.0f; }
    __syncthreads();

    // A-fragment (16x32 bf16, ISA §7.12.2): lane=M row; K octet interleave by lane-half.
    auto loadA = [&](const unsigned short* rowbase, int k0) -> v16bf {
        const unsigned int* s32 = (const unsigned int*)rowbase;
        v8u p;
        #pragma unroll
        for (int v = 0; v < 8; ++v) {
            int k = k0 + ((v & 3) << 1) + ((v >> 2) << 4) + (half << 3);
            p[v] = s32[k >> 1];
        }
        return __builtin_bit_cast(v16bf, p);
    };
    // B-fragment (32x16 bf16): lane=N column, K octet interleave by lane-half (same as A); Wt is [N][K] so
    // each (k,k+1) pair is one dword.
    auto loadB = [&](const unsigned short* Wt, int ldk, int n0, int k0) -> v16bf {
        const unsigned int* s32 = (const unsigned int*)(Wt + (n0 + mlan) * ldk);
        v8u p;
        #pragma unroll
        for (int v = 0; v < 8; ++v) {
            int k = k0 + ((v & 3) << 1) + ((v >> 2) << 4) + (half << 3);
            p[v] = s32[k >> 1];
        }
        return __builtin_bit_cast(v16bf, p);
    };
    auto mma = [&](v16bf a, v16bf b, v8f c) -> v8f {
        return __builtin_amdgcn_wmma_f32_16x16x32_bf16(false, a, false, b, (short)0, c, false, false);
    };
    // Elementwise LSTM cell; thread owns 8 fixed (m,h) cells -> c stays in VGPRs.
    auto cell = [&](float* cr, unsigned short* sH, bool save) {
        #pragma unroll
        for (int j = 0; j < 8; ++j) {
            int e  = tid * 8 + j;
            int m  = e >> 7;
            int hh = e & (H_DIM - 1);
            const float* gr = sG + m * G_DIM + hh;
            float gi = gr[0], gf = gr[H_DIM], gg = gr[2 * H_DIM], go = gr[3 * H_DIM];
            float c = sigmoidf(gf) * cr[j] + sigmoidf(gi) * tanhf(gg);
            float h = sigmoidf(go) * tanhf(c);
            cr[j] = c;
            sH[m * H_DIM + hh] = f2bf(h);
            if (save) h2f[j] = h;
        }
    };

    for (int t = 0; t < S_DIM; ++t) {
        // Optimization barrier: forbid LICM from hoisting the (loop-invariant)
        // weight/bias fragment loads out of the 1024-step loop -- hoisting them
        // exceeds the VGPR budget and turns them into scratch spill/reload,
        // which is strictly worse than re-reading LDS / L2-resident weights.
        asm volatile("" ::: "memory");

        // ---- Layer 1 gates: feats_t @ W_i1 + h1 @ W_h1 + (bi1+bh1) ----
        const unsigned short* frow = feats + ((m0 + mlan) * S_DIM + t) * CMID;
        if (t + 1 < S_DIM) __builtin_prefetch(frow + CMID, 0, 1);   // global_prefetch next step
        v16bf aF = loadA(frow, 0);
        v16bf aH1[4];
        #pragma unroll
        for (int kb = 0; kb < 4; ++kb) aH1[kb] = loadA(sH1 + mlan * H_DIM, kb * 32);

        #pragma unroll
        for (int tt = 0; tt < 4; ++tt) {
            int n0 = (wv * 4 + tt) * 16;
            float bia = sB1[n0 + mlan];
            v8f acc;
            #pragma unroll
            for (int r = 0; r < 8; ++r) acc[r] = bia;
            acc = mma(aF, loadB(Wti1, CMID, n0, 0), acc);
            #pragma unroll
            for (int kb = 0; kb < 4; ++kb) acc = mma(aH1[kb], loadB(sWh1, H_DIM, n0, kb * 32), acc);
            #pragma unroll
            for (int r = 0; r < 8; ++r) sG[(r + half * 8) * G_DIM + n0 + mlan] = acc[r];
        }
        __syncthreads();
        cell(c1r, sH1, false);
        __syncthreads();

        // ---- Layer 2 gates: h1' @ W_i2 + h2 @ W_h2 + (bi2+bh2) ----
        v16bf aH1n[4], aH2[4];
        #pragma unroll
        for (int kb = 0; kb < 4; ++kb) {
            aH1n[kb] = loadA(sH1 + mlan * H_DIM, kb * 32);
            aH2[kb]  = loadA(sH2 + mlan * H_DIM, kb * 32);
        }
        #pragma unroll
        for (int tt = 0; tt < 4; ++tt) {
            int n0 = (wv * 4 + tt) * 16;
            float bia = sB2[n0 + mlan];
            v8f acc;
            #pragma unroll
            for (int r = 0; r < 8; ++r) acc[r] = bia;
            #pragma unroll
            for (int kb = 0; kb < 4; ++kb) acc = mma(aH1n[kb], loadB(Wti2, H_DIM, n0, kb * 32), acc);
            #pragma unroll
            for (int kb = 0; kb < 4; ++kb) acc = mma(aH2[kb],  loadB(sWh2, H_DIM, n0, kb * 32), acc);
            #pragma unroll
            for (int r = 0; r < 8; ++r) sG[(r + half * 8) * G_DIM + n0 + mlan] = acc[r];
        }
        __syncthreads();
        cell(c2r, sH2, true);
        __syncthreads();
    }

    // ---- Final FC: out = h2 @ fc_w + fc_b (fp32 h2 staged through sG) ----
    #pragma unroll
    for (int j = 0; j < 8; ++j) {
        int e = tid * 8 + j;
        sG[(e >> 7) * G_DIM + (e & (H_DIM - 1))] = h2f[j];
    }
    __syncthreads();
    if (tid < BT * NCLS) {
        int m = tid / NCLS, cls = tid % NCLS;
        float a = fcb[cls];
        for (int h = 0; h < H_DIM; ++h) a += sG[m * G_DIM + h] * fcw[h * NCLS + cls];
        out[(m0 + m) * NCLS + cls] = a;
    }
}

// ---------------------------------------------------------------------------
// Host launch
// ---------------------------------------------------------------------------
extern "C" void kernel_launch(void* const* d_in, const int* in_sizes, int n_in,
                              void* d_out, int out_size, void* d_ws, size_t ws_size,
                              hipStream_t stream)
{
    (void)in_sizes; (void)n_in; (void)out_size; (void)ws_size;

    const float* x = (const float*)d_in[0];
    char* w = (char*)d_ws;
    // Workspace layout (bytes):
    float*          bufY    = (float*)(w);                              // 32 MB  pre-BN activations
    float*          bufA    = (float*)(w + (size_t)33554432);           // 32 MB  post-BN activations
    unsigned short* featsBf = (unsigned short*)(w + (size_t)67108864);  // 16 MB  bf16 features
    float*          partial = (float*)(w + (size_t)83886080);           // 256 KB BN partials
    float*          scsh    = (float*)(w + (size_t)84148224);           // 256 B  fused BN scale/shift
    unsigned short* Wti1    = (unsigned short*)(w + (size_t)84148480);  // 32 KB
    unsigned short* Wth1    = (unsigned short*)(w + (size_t)84181248);  // 128 KB
    unsigned short* Wti2    = (unsigned short*)(w + (size_t)84312320);  // 128 KB
    unsigned short* Wth2    = (unsigned short*)(w + (size_t)84443392);  // 128 KB
    float*          b1      = (float*)(w + (size_t)84574464);           // 2 KB
    float*          b2      = (float*)(w + (size_t)84576512);           // 2 KB

    const float inv_n = 1.0f / (float)(B_DIM * S_DIM);
    const int n_elem  = B_DIM * S_DIM * 32;

    for (int L = 0; L < 4; ++L) {
        const float* actin = (L == 0) ? x : bufA;
        int cin  = (L == 0) ? CIN0 : CMID;
        int base = 1 + 6 * L;   // dw_k, dw_b, pw_k, pw_b, bn_g, bn_b
        dwpw_kernel<<<(B_DIM * S_DIM) / 256, 256, 0, stream>>>(
            actin, (const float*)d_in[base], (const float*)d_in[base + 1],
            (const float*)d_in[base + 2], (const float*)d_in[base + 3],
            bufY, partial, cin, S_DIM);
        bnstats_kernel<<<1, 32, 0, stream>>>(
            partial, (B_DIM * S_DIM) / 256,
            (const float*)d_in[base + 4], (const float*)d_in[base + 5], scsh, inv_n);
        bnapply_kernel<<<n_elem / 256, 256, 0, stream>>>(
            bufY, scsh, bufA, (L == 3) ? featsBf : (unsigned short*)nullptr, n_elem);
    }

    weight_prep_kernel<<<834, 256, 0, stream>>>(
        (const float*)d_in[25], (const float*)d_in[26], (const float*)d_in[27], (const float*)d_in[28],
        (const float*)d_in[29], (const float*)d_in[30],
        (const float*)d_in[31], (const float*)d_in[32], (const float*)d_in[33], (const float*)d_in[34],
        (const float*)d_in[35], (const float*)d_in[36],
        Wti1, Wth1, Wti2, Wth2, b1, b2);

    // LDS: Wh1 + Wh2 (256 KB) + gates (32 KB) + h1/h2 (8 KB) + biases (4 KB)
    const int lstm_lds = 2 * G_DIM * H_DIM * 2 + BT * G_DIM * 4 + 2 * BT * H_DIM * 2
                       + 2 * G_DIM * 4;                                  // 307200 B
    (void)hipFuncSetAttribute(reinterpret_cast<const void*>(lstm_kernel),
                              hipFuncAttributeMaxDynamicSharedMemorySize, lstm_lds);
    lstm_kernel<<<B_DIM / BT, 256, lstm_lds, stream>>>(
        featsBf, Wti1, Wth1, Wti2, Wth2, b1, b2,
        (const float*)d_in[37], (const float*)d_in[38], (float*)d_out);
}
